// LinearAttention1D_42356967473203
// MI455X (gfx1250) — matrix operand order repaired and checked by reading, not compiled, not run on hardware
//
#include <hip/hip_runtime.h>
#include <hip/hip_bf16.h>
#include <math.h>

// ---------------------------------------------------------------------------
// LinearAttention1D fused pipeline for gfx1250 (MI455X), wave32 WMMA bf16.
//   B=16, C=512, N=4096, H=8, D=64.  All GEMMs -> v_wmma_f32_16x16x32_bf16.
//   Round 2: double-buffered B fragments + 8x B-fragment reuse in K3.
// ---------------------------------------------------------------------------

typedef __attribute__((ext_vector_type(16))) __bf16 v16bf;
typedef __attribute__((ext_vector_type(8)))  float  v8f;

#define B_   16
#define C_   512
#define N_   4096
#define H_   8
#define NEG_BIG (-3.0e38f)

// workspace layout (bytes); total ~10.5 MB
#define OFF_WQKV 0
#define SZ_WQKV  (1536 * 512 * 2)
#define OFF_WOUT (OFF_WQKV + SZ_WQKV)
#define SZ_WOUT  (512 * 512 * 2)
#define OFF_CTXP (OFF_WOUT + SZ_WOUT)            // 4 chunks x 128 bh x 64x64 f32
#define SZ_CTXP  (4 * 128 * 4096 * 4)
#define OFF_KST  (OFF_CTXP + SZ_CTXP)            // 4 chunks x 128 bh x 64 x {m,s}

// --------------------------- WMMA fragment helpers -------------------------
// A fragment (16x32, bf16): element (row,k) at base[row*ld + k]
// lanes 0-15 K={0..7,16..23}, lanes 16-31 K={8..15,24..31} (ISA 7.12.2)
template <typename T>
static __device__ __forceinline__ v16bf load_fragA(const T* base, int ld) {
  const int lane = threadIdx.x & 31;
  const int row = lane & 15, hi = lane >> 4;
  v16bf f;
#pragma unroll
  for (int p = 0; p < 8; ++p) {
    const int k0 = (p < 4 ? 0 : 8) + hi * 8 + 2 * p;
    const T* s = base + row * ld + k0;
    f[2 * p]     = (__bf16)s[0];
    f[2 * p + 1] = (__bf16)s[1];
  }
  return f;
}

// A fragment built from 4 fp32 partial context buffers (summed, then bf16)
static __device__ __forceinline__ v16bf load_fragA_ctx4(const float* base, int ld,
                                                        long cstride) {
  const int lane = threadIdx.x & 31;
  const int row = lane & 15, hi = lane >> 4;
  v16bf f;
#pragma unroll
  for (int p = 0; p < 8; ++p) {
    const int k0 = (p < 4 ? 0 : 8) + hi * 8 + 2 * p;
    const float* s = base + row * ld + k0;
#pragma unroll
    for (int e = 0; e < 2; ++e) {
      float v = s[e] + s[e + cstride] + s[e + 2 * cstride] + s[e + 3 * cstride];
      f[2 * p + e] = (__bf16)v;
    }
  }
  return f;
}

// B fragment (32x16, bf16): element (k,col) at base[col*ld + k]
// lanes 0-15 K={0..7,16..23}, lanes 16-31 K={8..15,24..31} (matches A fragment)
static __device__ __forceinline__ v16bf load_fragBT(const __bf16* base, int ld) {
  const int lane = threadIdx.x & 31;
  const int col = lane & 15, hi = lane >> 4;
  v16bf f;
#pragma unroll
  for (int p = 0; p < 8; ++p) {
    const int k0 = (p < 4 ? 0 : 8) + hi * 8 + 2 * p;
    const __bf16* s = base + col * ld + k0;
    f[2 * p]     = s[0];
    f[2 * p + 1] = s[1];
  }
  return f;
}

static __device__ __forceinline__ v8f wmma_bf16(v16bf a, v16bf b, v8f c) {
  return __builtin_amdgcn_wmma_f32_16x16x32_bf16(false, a, false, b, (short)0, c,
                                                 false, false);
}

#define V8F_ZERO ((v8f){0, 0, 0, 0, 0, 0, 0, 0})

// ----------------------------- K0: weight convert --------------------------
__global__ void la_k0_cvt(const float* __restrict__ wqkv, const float* __restrict__ wout,
                          __bf16* __restrict__ wqkv_bf, __bf16* __restrict__ wout_bf) {
  const int nq = 1536 * 512, no = 512 * 512;
  for (int i = blockIdx.x * blockDim.x + threadIdx.x; i < nq + no;
       i += gridDim.x * blockDim.x) {
    if (i < nq) wqkv_bf[i] = (__bf16)wqkv[i];
    else        wout_bf[i - nq] = (__bf16)wout[i - nq];
  }
}

// ----------------------- K1: k softmax stats (online) -----------------------
// grid = 128 (b,h) * 4 n-chunks; block = 128 (4 waves). Wave w owns k rows 16w..16w+15.
__global__ __launch_bounds__(128)
void la_k1_kstats(const float* __restrict__ x, const __bf16* __restrict__ wqkv_bf,
                  float* __restrict__ kstats) {
  __shared__ __bf16 x_l[16 * 520];
  const int bh = blockIdx.x >> 2, chunk = blockIdx.x & 3;
  const int b = bh >> 3, h = bh & 7;
  const int tid = threadIdx.x, wid = tid >> 5, lane = tid & 31;
  const int hi = lane >> 4;

  // Preload this wave's 16 k-weight rows as A fragments (16 K-chunks of 32)
  const __bf16* wk = wqkv_bf + (size_t)(512 + h * 64 + wid * 16) * 512;
  v16bf Ak[16];
#pragma unroll
  for (int kc = 0; kc < 16; ++kc) Ak[kc] = load_fragA(wk + kc * 32, 512);

  float m[8], s[8];
#pragma unroll
  for (int r = 0; r < 8; ++r) { m[r] = NEG_BIG; s[r] = 0.f; }

  const float* xb = x + (size_t)b * C_ * N_;
  for (int nt = 0; nt < 64; ++nt) {
    const int n0 = chunk * 1024 + nt * 16;
    __syncthreads();
    for (int idx = tid; idx < C_ * 16; idx += 128) {
      int c = idx >> 4, n = idx & 15;
      x_l[n * 520 + c] = (__bf16)xb[(size_t)c * N_ + n0 + n];
    }
    __syncthreads();
    v8f acc = V8F_ZERO;
    v16bf bT = load_fragBT(x_l, 520);  // double-buffered B fragment
#pragma unroll
    for (int kc = 0; kc < 16; ++kc) {
      v16bf bN = (kc < 15) ? load_fragBT(x_l + (kc + 1) * 32, 520) : bT;
      acc = wmma_bf16(Ak[kc], bT, acc);
      bT = bN;
    }
    // online max / sum-exp per row (reduce across 16 columns = 16-lane groups)
#pragma unroll
    for (int r = 0; r < 8; ++r) {
      float v = acc[r];
      float tm = v;
      tm = fmaxf(tm, __shfl_xor(tm, 1, 16));
      tm = fmaxf(tm, __shfl_xor(tm, 2, 16));
      tm = fmaxf(tm, __shfl_xor(tm, 4, 16));
      tm = fmaxf(tm, __shfl_xor(tm, 8, 16));
      float nm = fmaxf(m[r], tm);
      float p = __expf(v - nm);
      p += __shfl_xor(p, 1, 16);
      p += __shfl_xor(p, 2, 16);
      p += __shfl_xor(p, 4, 16);
      p += __shfl_xor(p, 8, 16);
      s[r] = s[r] * __expf(m[r] - nm) + p;
      m[r] = nm;
    }
  }
  if ((lane & 15) == 0) {
#pragma unroll
    for (int r = 0; r < 8; ++r) {
      const int row = wid * 16 + r + hi * 8;
      float* o = kstats + ((size_t)(chunk * 128 + bh) * 64 + row) * 2;
      o[0] = m[r];
      o[1] = s[r];
    }
  }
}

// ------------------------- K2: context = knorm x v^T -------------------------
// grid = 128 (b,h) * 4 chunks; block = 128. Wave w owns k/v rows 16w..16w+15,
// and context rows 16w..16w+15 (all 64 columns).
__global__ __launch_bounds__(128)
void la_k2_context(const float* __restrict__ x, const __bf16* __restrict__ wqkv_bf,
                   const float* __restrict__ kstats, float* __restrict__ ctxp) {
  __shared__ __bf16 x_l[32 * 520];
  __shared__ __bf16 k_l[64 * 40];
  __shared__ __bf16 v_l[64 * 40];
  const int bh = blockIdx.x >> 2, chunk = blockIdx.x & 3;
  const int b = bh >> 3, h = bh & 7;
  const int tid = threadIdx.x, wid = tid >> 5, lane = tid & 31;
  const int hi = lane >> 4, col = lane & 15;
  const int rb = wid * 16;

  const __bf16* wk = wqkv_bf + (size_t)(512 + h * 64 + rb) * 512;
  const __bf16* wv = wqkv_bf + (size_t)(1024 + h * 64 + rb) * 512;
  v16bf Ak[16], Av[16];
#pragma unroll
  for (int kc = 0; kc < 16; ++kc) {
    Ak[kc] = load_fragA(wk + kc * 32, 512);
    Av[kc] = load_fragA(wv + kc * 32, 512);
  }

  // merge the 4 partial softmax stats for this wave's rows
  float m[8], sinv[8];
#pragma unroll
  for (int r = 0; r < 8; ++r) {
    const int row = rb + r + hi * 8;
    float pm[4], ps[4], M = NEG_BIG;
#pragma unroll
    for (int c = 0; c < 4; ++c) {
      const float* st = kstats + ((size_t)(c * 128 + bh) * 64 + row) * 2;
      pm[c] = st[0]; ps[c] = st[1];
      M = fmaxf(M, pm[c]);
    }
    float S = 0.f;
#pragma unroll
    for (int c = 0; c < 4; ++c) S += ps[c] * __expf(pm[c] - M);
    m[r] = M;
    sinv[r] = 1.0f / S;
  }

  v8f cacc[4];
#pragma unroll
  for (int jt = 0; jt < 4; ++jt) cacc[jt] = V8F_ZERO;

  const float* xb = x + (size_t)b * C_ * N_;
  for (int nt = 0; nt < 32; ++nt) {
    const int n0 = chunk * 1024 + nt * 32;
    __syncthreads();
    for (int idx = tid; idx < C_ * 32; idx += 128) {
      int c = idx >> 5, n = idx & 31;
      x_l[n * 520 + c] = (__bf16)xb[(size_t)c * N_ + n0 + n];
    }
    __syncthreads();
#pragma unroll
    for (int t = 0; t < 2; ++t) {
      v8f ak = V8F_ZERO;
      v8f av = V8F_ZERO;
      v16bf bT = load_fragBT(x_l + t * 16 * 520, 520);  // double-buffered
#pragma unroll
      for (int kc = 0; kc < 16; ++kc) {
        v16bf bN = (kc < 15) ? load_fragBT(x_l + t * 16 * 520 + (kc + 1) * 32, 520) : bT;
        ak = wmma_bf16(Ak[kc], bT, ak);   // 2 WMMAs per B fragment
        av = wmma_bf16(Av[kc], bT, av);
        bT = bN;
      }
#pragma unroll
      for (int r = 0; r < 8; ++r) {
        const int i = rb + r + hi * 8;
        const int nl = t * 16 + col;
        k_l[i * 40 + nl] = (__bf16)(__expf(ak[r] - m[r]) * sinv[r]);
        v_l[i * 40 + nl] = (__bf16)av[r];
      }
    }
    __syncthreads();
    const v16bf aK = load_fragA((const __bf16*)k_l + rb * 40, 40);
    v16bf bV = load_fragBT(v_l, 40);  // double-buffered
#pragma unroll
    for (int jt = 0; jt < 4; ++jt) {
      v16bf bn = (jt < 3) ? load_fragBT(v_l + (jt + 1) * 16 * 40, 40) : bV;
      cacc[jt] = wmma_bf16(aK, bV, cacc[jt]);
      bV = bn;
    }
  }
  // store partial context transposed: [j][i], i contiguous
  float* cp = ctxp + (size_t)(chunk * 128 + bh) * 4096;
#pragma unroll
  for (int jt = 0; jt < 4; ++jt)
#pragma unroll
    for (int r = 0; r < 8; ++r) {
      const int j = jt * 16 + col;
      const int i = rb + r + hi * 8;
      cp[j * 64 + i] = cacc[jt][r];
    }
}

// --------- K3: q softmax, ctx^T x q, out projection, bias, LayerNorm ---------
// grid = 16 b * 256 n-tiles (16 cols); block = 128. Wave w owns rows 128w..128w+127.
// GEMMs are kc-outer / row-tile-inner: one B fragment feeds 8 WMMAs.
__global__ __launch_bounds__(128)
void la_k3_output(const float* __restrict__ x, const __bf16* __restrict__ wqkv_bf,
                  const __bf16* __restrict__ wout_bf, const float* __restrict__ ctxp,
                  const float* __restrict__ b_out, const float* __restrict__ g_out,
                  float* __restrict__ out) {
  __shared__ __bf16 x_l[16 * 520];
  __shared__ __bf16 q_l[16 * 520];
  __shared__ __bf16 o_l[16 * 520];
  __shared__ float red[4][2][16];
  const int b = blockIdx.x >> 8, ntile = blockIdx.x & 255;
  const int n0 = ntile * 16;
  const int tid = threadIdx.x, wid = tid >> 5, lane = tid & 31;
  const int hi = lane >> 4, col = lane & 15;

  const float* xb = x + (size_t)b * C_ * N_;
  for (int idx = tid; idx < C_ * 16; idx += 128) {
    int c = idx >> 4, n = idx & 15;
    x_l[n * 520 + c] = (__bf16)xb[(size_t)c * N_ + n0 + n];
  }
  __syncthreads();

  // ---- q projection: 8 row-tiles share each B fragment ----
  v8f qacc[8];
#pragma unroll
  for (int rt = 0; rt < 8; ++rt) qacc[rt] = V8F_ZERO;
  {
    v16bf bT = load_fragBT(x_l, 520);
#pragma unroll
    for (int kc = 0; kc < 16; ++kc) {
      v16bf bN = (kc < 15) ? load_fragBT(x_l + (kc + 1) * 32, 520) : bT;
#pragma unroll
      for (int rt = 0; rt < 8; ++rt) {
        const __bf16* wq = wqkv_bf + (size_t)(wid * 128 + rt * 16) * 512 + kc * 32;
        qacc[rt] = wmma_bf16(load_fragA(wq, 512), bT, qacc[rt]);
      }
      bT = bN;
    }
  }
  float qv[8][8];
#pragma unroll
  for (int rt = 0; rt < 8; ++rt)
#pragma unroll
    for (int r = 0; r < 8; ++r) qv[rt][r] = qacc[rt][r];

  // ---- softmax over head dim (64 rows) per column; heads 2w, 2w+1 ----
#pragma unroll
  for (int hh = 0; hh < 2; ++hh) {
    float mx = NEG_BIG;
#pragma unroll
    for (int rt = hh * 4; rt < hh * 4 + 4; ++rt)
#pragma unroll
      for (int r = 0; r < 8; ++r) mx = fmaxf(mx, qv[rt][r]);
    mx = fmaxf(mx, __shfl_xor(mx, 16, 32));
    float sm = 0.f;
#pragma unroll
    for (int rt = hh * 4; rt < hh * 4 + 4; ++rt)
#pragma unroll
      for (int r = 0; r < 8; ++r) sm += __expf(qv[rt][r] - mx);
    sm += __shfl_xor(sm, 16, 32);
    const float inv = 0.125f / sm;  // scale = DIM_HEAD^-0.5
#pragma unroll
    for (int rt = hh * 4; rt < hh * 4 + 4; ++rt)
#pragma unroll
      for (int r = 0; r < 8; ++r) {
        const int row = wid * 128 + rt * 16 + r + hi * 8;
        q_l[col * 520 + row] = (__bf16)(__expf(qv[rt][r] - mx) * inv);
      }
  }
  __syncthreads();

  // ---- apply context: out[j,n] = sum_i ctx^T[j,i] * q[i,n] (per head) ----
  // 4 distinct B fragments, each loaded once and feeding 4 WMMAs.
  v8f oacc[8];
#pragma unroll
  for (int rt = 0; rt < 8; ++rt) oacc[rt] = V8F_ZERO;
#pragma unroll
  for (int hh = 0; hh < 2; ++hh) {
    const int hg = wid * 2 + hh;  // head handled by rows (wid*128 + hh*64 ..)
#pragma unroll
    for (int ic = 0; ic < 2; ++ic) {
      v16bf bb = load_fragBT(q_l + hg * 64 + ic * 32, 520);
#pragma unroll
      for (int t = 0; t < 4; ++t) {
        const int rt = hh * 4 + t;
        v16bf a = load_fragA_ctx4(
            ctxp + (size_t)(b * 8 + hg) * 4096 + t * 16 * 64 + ic * 32, 64,
            (long)128 * 4096);
        oacc[rt] = wmma_bf16(a, bb, oacc[rt]);
      }
    }
  }
#pragma unroll
  for (int rt = 0; rt < 8; ++rt)
#pragma unroll
    for (int r = 0; r < 8; ++r) {
      const int row = wid * 128 + rt * 16 + r + hi * 8;
      o_l[col * 520 + row] = (__bf16)oacc[rt][r];
    }
  __syncthreads();

  // ---- output projection (8 row-tiles per B fragment) + bias ----
  v8f yacc[8];
#pragma unroll
  for (int rt = 0; rt < 8; ++rt) yacc[rt] = V8F_ZERO;
  {
    v16bf bT = load_fragBT(o_l, 520);
#pragma unroll
    for (int kc = 0; kc < 16; ++kc) {
      v16bf bN = (kc < 15) ? load_fragBT(o_l + (kc + 1) * 32, 520) : bT;
#pragma unroll
      for (int rt = 0; rt < 8; ++rt) {
        const __bf16* wo = wout_bf + (size_t)(wid * 128 + rt * 16) * 512 + kc * 32;
        yacc[rt] = wmma_bf16(load_fragA(wo, 512), bT, yacc[rt]);
      }
      bT = bN;
    }
  }
  float yv[8][8];
#pragma unroll
  for (int rt = 0; rt < 8; ++rt)
#pragma unroll
    for (int r = 0; r < 8; ++r) {
      const int row = wid * 128 + rt * 16 + r + hi * 8;
      yv[rt][r] = yacc[rt][r] + b_out[row];
    }

  // ---- channel LayerNorm over 512 rows per column ----
  float ps = 0.f, ps2 = 0.f;
#pragma unroll
  for (int rt = 0; rt < 8; ++rt)
#pragma unroll
    for (int r = 0; r < 8; ++r) { float v = yv[rt][r]; ps += v; ps2 += v * v; }
  ps  += __shfl_xor(ps, 16, 32);
  ps2 += __shfl_xor(ps2, 16, 32);
  if (hi == 0) { red[wid][0][col] = ps; red[wid][1][col] = ps2; }
  __syncthreads();
  const float tot  = red[0][0][col] + red[1][0][col] + red[2][0][col] + red[3][0][col];
  const float tot2 = red[0][1][col] + red[1][1][col] + red[2][1][col] + red[3][1][col];
  const float mean = tot * (1.0f / 512.0f);
  const float var  = tot2 * (1.0f / 512.0f) - mean * mean;
  const float rstd = rsqrtf(var + 1e-5f);
  float* ob = out + (size_t)b * 512 * N_ + n0;
#pragma unroll
  for (int rt = 0; rt < 8; ++rt)
#pragma unroll
    for (int r = 0; r < 8; ++r) {
      const int row = wid * 128 + rt * 16 + r + hi * 8;
      ob[(size_t)row * N_ + col] = (yv[rt][r] - mean) * rstd * g_out[row];
    }
}

// --------------------------------- launch ----------------------------------
extern "C" void kernel_launch(void* const* d_in, const int* in_sizes, int n_in,
                              void* d_out, int out_size, void* d_ws, size_t ws_size,
                              hipStream_t stream) {
  const float* x     = (const float*)d_in[0];
  const float* w_qkv = (const float*)d_in[1];
  const float* w_out = (const float*)d_in[2];
  const float* b_out = (const float*)d_in[3];
  const float* g_out = (const float*)d_in[4];
  float* out = (float*)d_out;

  char* ws = (char*)d_ws;
  __bf16* wqkv_bf = (__bf16*)(ws + OFF_WQKV);
  __bf16* wout_bf = (__bf16*)(ws + OFF_WOUT);
  float*  ctxp    = (float*)(ws + OFF_CTXP);
  float*  kstats  = (float*)(ws + OFF_KST);

  la_k0_cvt<<<4096, 256, 0, stream>>>(w_qkv, w_out, wqkv_bf, wout_bf);
  la_k1_kstats<<<512, 128, 0, stream>>>(x, wqkv_bf, kstats);
  la_k2_context<<<512, 128, 0, stream>>>(x, wqkv_bf, kstats, ctxp);
  la_k3_output<<<4096, 128, 0, stream>>>(x, wqkv_bf, wout_bf, ctxp, b_out, g_out, out);
}
